// DeltaNet_31877247271574
// MI455X (gfx1250) — matrix operand order repaired and checked by reading, not compiled, not run on hardware
//
#include <hip/hip_runtime.h>
#include <hip/hip_bf16.h>
#include <cstdint>

typedef _Float16 f16;
typedef __attribute__((ext_vector_type(16))) _Float16 v16h;
typedef __attribute__((ext_vector_type(8)))  float    v8f;

#define B_   2
#define L_   4096
#define D_   1024
#define H_   4
#define DK_  256
#define DV_  256
#define CS_  32
#define NC_  128            /* L_/CS_ */
#define BL_  (B_*L_)        /* 8192   */
#define GIN_ 1792
#define GH_  1024
#define NSLICE_ 4           /* dv slices for delta scan */
#define SLICE_  64          /* DV_/NSLICE_ */

/* ---------------- fragment helpers (16-bit A/B, contiguous k = 16h + s per lane half) ------ */

__device__ __forceinline__ v16h ld_frag(const f16* p0, const f16* p1) {
  union { uint4 u[2]; v16h v; } t;
  t.u[0] = *(const uint4*)p0;
  t.u[1] = *(const uint4*)p1;
  return t.v;
}
/* load + negate (flip f16 sign bits): fp16 WMMA has no A/B NEG modifier */
__device__ __forceinline__ v16h ld_frag_neg(const f16* p0, const f16* p1) {
  uint4 a = *(const uint4*)p0, b = *(const uint4*)p1;
  a.x ^= 0x80008000u; a.y ^= 0x80008000u; a.z ^= 0x80008000u; a.w ^= 0x80008000u;
  b.x ^= 0x80008000u; b.y ^= 0x80008000u; b.z ^= 0x80008000u; b.w ^= 0x80008000u;
  union { uint4 u[2]; v16h v; } t; t.u[0] = a; t.u[1] = b; return t.v;
}

#define WMMA(Afr, Bfr, Cfr) \
  __builtin_amdgcn_wmma_f32_16x16x32_f16(false, (Afr), false, (Bfr), (short)0, (Cfr), false, false)

/* ---------------- f32 -> f16 conversion ----------------------------------- */

__global__ __launch_bounds__(256) void cvt_f16_kernel(const float* __restrict__ in,
                                                      f16* __restrict__ out, size_t n) {
  size_t i = (size_t)blockIdx.x * 256 + threadIdx.x;
  if (i < n) out[i] = (f16)in[i];
}

/* ---------------- big GEMM: C[M,N] = A[M,K](f16) x W[N,K]^T(f16) ---------- */
/* block = 128x128 tile, K step 32; 8 waves in 4(M)x2(N) grid, each wave 32x64 */

#define LDT 40   /* padded LDS row stride (halves) */

__global__ __launch_bounds__(256) void gemm_f16_kernel(
    const f16* __restrict__ A, const f16* __restrict__ Bw,
    float* __restrict__ C, int M, int N, int K) {
  (void)M;
  __shared__ __align__(16) f16 sA[128 * LDT];
  __shared__ __align__(16) f16 sB[128 * LDT];
  const int tid  = threadIdx.x;
  const int lane = tid & 31;
  const int wv   = tid >> 5;
  const int wm   = wv & 3;        /* 0..3 : 32-row band   */
  const int wn   = wv >> 2;       /* 0..1 : 64-col band   */
  const int m0   = blockIdx.y * 128;
  const int n0   = blockIdx.x * 128;
  const int lr   = lane & 15;
  const int hi   = lane >> 4;

  v8f acc[2][4];
#pragma unroll
  for (int i = 0; i < 2; ++i)
#pragma unroll
    for (int j = 0; j < 4; ++j) acc[i][j] = {};

  const int ldr = tid >> 1;           /* 0..127 */
  const int ldc = (tid & 1) * 16;     /* 0 / 16 */

  for (int k0 = 0; k0 < K; k0 += 32) {
    {
      const f16* sa = A  + (size_t)(m0 + ldr) * K + k0 + ldc;
      const f16* sb = Bw + (size_t)(n0 + ldr) * K + k0 + ldc;
      *(uint4*)&sA[ldr * LDT + ldc]     = *(const uint4*)(sa);
      *(uint4*)&sA[ldr * LDT + ldc + 8] = *(const uint4*)(sa + 8);
      *(uint4*)&sB[ldr * LDT + ldc]     = *(const uint4*)(sb);
      *(uint4*)&sB[ldr * LDT + ldc + 8] = *(const uint4*)(sb + 8);
    }
    if (k0 + 32 < K) {  /* prefetch next K tile -> global_prefetch_b8 */
      __builtin_prefetch(A  + (size_t)(m0 + ldr) * K + k0 + 32 + ldc, 0, 1);
      __builtin_prefetch(Bw + (size_t)(n0 + ldr) * K + k0 + 32 + ldc, 0, 1);
    }
    __syncthreads();

    v16h afr[2], bfr[4];
#pragma unroll
    for (int mt = 0; mt < 2; ++mt) {
      const f16* p = &sA[(wm * 32 + mt * 16 + lr) * LDT + hi * 16];
      afr[mt] = ld_frag(p, p + 8);    /* K contiguous 16 halves         */
    }
#pragma unroll
    for (int nt = 0; nt < 4; ++nt) {
      const f16* p = &sB[(wn * 64 + nt * 16 + lr) * LDT + hi * 16];
      bfr[nt] = ld_frag(p, p + 8);    /* K contiguous 16 halves         */
    }
#pragma unroll
    for (int mt = 0; mt < 2; ++mt)
#pragma unroll
      for (int nt = 0; nt < 4; ++nt)
        acc[mt][nt] = WMMA(afr[mt], bfr[nt], acc[mt][nt]);
    __syncthreads();
  }

#pragma unroll
  for (int mt = 0; mt < 2; ++mt)
#pragma unroll
    for (int nt = 0; nt < 4; ++nt) {
      const int n = n0 + wn * 64 + nt * 16 + lr;
      const int mb = m0 + wm * 32 + mt * 16 + hi * 8;
#pragma unroll
      for (int r = 0; r < 8; ++r)
        C[(size_t)(mb + r) * N + n] = acc[mt][nt][r];
    }
}

/* ---------------- beta = sigmoid(hs @ b_proj^T)  (N=4, tiny) -------------- */

__global__ __launch_bounds__(256) void bproj_kernel(const float* __restrict__ hs,
                                                    const float* __restrict__ Wb,
                                                    float* __restrict__ beta) {
  int t = blockIdx.x * 256 + threadIdx.x;       /* BL_*4 threads */
  int h = t & 3;
  size_t bl = (size_t)(t >> 2);
  const float* x = hs + bl * D_;
  const float* w = Wb + h * D_;
  float acc = 0.f;
  for (int k = 0; k < D_; ++k) acc += x[k] * w[k];
  beta[(size_t)t] = 1.f / (1.f + expf(-acc));
}

/* ---------------- causal depthwise conv (k=4) + SiLU ---------------------- */

__global__ __launch_bounds__(256) void conv4_silu_kernel(const float* __restrict__ x,
                                                         const float* __restrict__ w,
                                                         float* __restrict__ y) {
  size_t i = (size_t)blockIdx.x * 256 + threadIdx.x;   /* BL_*1024 */
  int c = (int)(i & 1023);
  size_t bl = i >> 10;
  int l = (int)(bl & (L_ - 1));
  size_t row0 = bl - l;
  float acc = 0.f;
#pragma unroll
  for (int j = 0; j < 4; ++j) {
    int ll = l - 3 + j;
    if (ll >= 0) acc += x[(row0 + ll) * 1024 + c] * w[c * 4 + j];
  }
  y[i] = acc / (1.f + expf(-acc));   /* silu */
}

/* ---------------- delta-rule pre: l2norm q,k; v*beta; kb=k̂*beta ----------- */

__global__ __launch_bounds__(256) void delta_pre_kernel(
    const float* __restrict__ qc, const float* __restrict__ kc,
    const float* __restrict__ vc, const float* __restrict__ beta,
    f16* __restrict__ qh, f16* __restrict__ kh, f16* __restrict__ kb,
    float* __restrict__ vb) {
  const int bl = blockIdx.x;
  const int t  = threadIdx.x;
  const int h  = t >> 6, j = t & 63;
  __shared__ float red[256];
  const size_t ib = (size_t)bl * 1024 + h * 256 + j * 4;
  float q0=qc[ib],q1=qc[ib+1],q2=qc[ib+2],q3=qc[ib+3];
  float k0=kc[ib],k1=kc[ib+1],k2=kc[ib+2],k3=kc[ib+3];
  float v0=vc[ib],v1=vc[ib+1],v2=vc[ib+2],v3=vc[ib+3];
  red[t] = q0*q0+q1*q1+q2*q2+q3*q3;
  __syncthreads();
  for (int off = 32; off >= 1; off >>= 1) { if (j < off) red[t] += red[t+off]; __syncthreads(); }
  float rq = rsqrtf(red[h*64] + 1e-12f);
  __syncthreads();
  red[t] = k0*k0+k1*k1+k2*k2+k3*k3;
  __syncthreads();
  for (int off = 32; off >= 1; off >>= 1) { if (j < off) red[t] += red[t+off]; __syncthreads(); }
  float rk = rsqrtf(red[h*64] + 1e-12f);
  const int b = bl / L_, l = bl % L_;
  const float bta = beta[(size_t)bl * 4 + h];
  const size_t ob = (((size_t)(b * H_ + h)) * L_ + l) * 256 + j * 4;
  qh[ob  ]=(f16)(q0*rq); qh[ob+1]=(f16)(q1*rq); qh[ob+2]=(f16)(q2*rq); qh[ob+3]=(f16)(q3*rq);
  kh[ob  ]=(f16)(k0*rk); kh[ob+1]=(f16)(k1*rk); kh[ob+2]=(f16)(k2*rk); kh[ob+3]=(f16)(k3*rk);
  kb[ob  ]=(f16)(k0*rk*bta); kb[ob+1]=(f16)(k1*rk*bta); kb[ob+2]=(f16)(k2*rk*bta); kb[ob+3]=(f16)(k3*rk*bta);
  vb[ob  ]=v0*bta; vb[ob+1]=v1*bta; vb[ob+2]=v2*bta; vb[ob+3]=v3*bta;
}

/* ---------------- per-chunk: A=strict_lo(kb k̂ᵀ); solve (I+A)u=vb,(I+A)w=kb  */
/* one block (128 thr / 4 waves) per (b,h,chunk); WMMA for the 32x32 Gram.    */

__global__ __launch_bounds__(128) void chunk_prep_kernel(
    const f16* __restrict__ khG, const f16* __restrict__ kbG,
    const float* __restrict__ vbG, float* __restrict__ uG, f16* __restrict__ wG) {
  extern __shared__ char smem_cp[];
  float* sA = (float*)smem_cp;                 /* 32*32        */
  float* sU = sA + 32 * 32;                    /* 32*256       */
  float* sW = sU + 32 * 256;                   /* 32*256       */
  const int blk = blockIdx.x;
  const int chunk = blk % NC_;
  const int bh    = blk / NC_;
  const size_t base = ((size_t)bh * L_ + (size_t)chunk * CS_) * DK_;
  const int t = threadIdx.x;
  const int lane = t & 31, wv = t >> 5;
  const int lr = lane & 15, hi = lane >> 4;
  const int mt = wv >> 1, nt = wv & 1;

  /* A tile via WMMA: A[i,j] = sum_d kb[i,d]*kh[j,d] */
  v8f acc = {};
#pragma unroll
  for (int kk = 0; kk < 8; ++kk) {
    const f16* ar = kbG + base + (size_t)(mt * 16 + lr) * DK_ + kk * 32;
    v16h af = ld_frag(ar + hi * 16, ar + hi * 16 + 8);
    const f16* br = khG + base + (size_t)(nt * 16 + lr) * DK_ + kk * 32 + hi * 16;
    v16h bf = ld_frag(br, br + 8);
    acc = WMMA(af, bf, acc);
  }
#pragma unroll
  for (int r = 0; r < 8; ++r) {
    int Mr = mt * 16 + hi * 8 + r, Nc = nt * 16 + lr;
    sA[Mr * 32 + Nc] = (Mr > Nc) ? acc[r] : 0.f;   /* strict lower */
  }
  /* rhs init (coalesced) */
  for (int idx = t; idx < 32 * 256; idx += 128) {
    sU[idx] = vbG[base + idx];
    sW[idx] = (float)kbG[base + idx];
  }
  __syncthreads();

  /* forward substitution, column-parallel (thread owns 2 cols of u and w) */
  const int c0 = 2 * t, c1 = 2 * t + 1;
  for (int i = 1; i < 32; ++i) {
    float a0 = sU[i * 256 + c0], a1 = sU[i * 256 + c1];
    float b0 = sW[i * 256 + c0], b1 = sW[i * 256 + c1];
    for (int jj = 0; jj < i; ++jj) {
      float a = sA[i * 32 + jj];
      a0 -= a * sU[jj * 256 + c0]; a1 -= a * sU[jj * 256 + c1];
      b0 -= a * sW[jj * 256 + c0]; b1 -= a * sW[jj * 256 + c1];
    }
    sU[i * 256 + c0] = a0; sU[i * 256 + c1] = a1;
    sW[i * 256 + c0] = b0; sW[i * 256 + c1] = b1;
  }
  __syncthreads();
  for (int idx = t; idx < 32 * 256; idx += 128) {
    uG[base + idx] = sU[idx];
    wG[base + idx] = (f16)sW[idx];
  }
}

/* ---------------- delta scan: sequential over 128 chunks ------------------ */
/* grid = B*H*NSLICE_ (32) blocks of 256 thr; S slice [256][64] f32 in LDS.   */

__global__ __launch_bounds__(256) void delta_scan_kernel(
    const f16* __restrict__ qG, const f16* __restrict__ kG,
    const float* __restrict__ uG, const f16* __restrict__ wG,
    float* __restrict__ deltaO) {
  extern __shared__ char smem_ds[];
  float* S  = (float*)smem_ds;                       /* 256*64 f32 = 64KB */
  f16*  kt  = (f16*)(smem_ds + 256 * SLICE_ * 4);    /* 32*256 f16        */
  f16*  up  = kt + 32 * 256;                         /* 32*64  f16        */
  f16*  at  = up + 32 * SLICE_;                      /* 32*32  f16        */

  const int bh    = blockIdx.x >> 2;
  const int slice = blockIdx.x & 3;
  const int n0    = slice * SLICE_;
  const int tid = threadIdx.x, lane = tid & 31, wv = tid >> 5;
  const int lr = lane & 15, hi = lane >> 4;
  const int mt = wv >> 2, nt = wv & 3;     /* step1/3 tile: 2(M) x 4(N) */
  const int b = bh >> 2, h = bh & 3;

  for (int i = tid; i < 256 * SLICE_; i += 256) S[i] = 0.f;
  __syncthreads();

  for (int c = 0; c < NC_; ++c) {
    const size_t cb = ((size_t)bh * L_ + (size_t)c * CS_) * DK_;

    /* stage k chunk (32x256 f16) into LDS */
    {
      const uint4* s4 = (const uint4*)(kG + cb + (size_t)tid * 32);
      uint4* d4 = (uint4*)(kt + tid * 32);
      d4[0] = s4[0]; d4[1] = s4[1]; d4[2] = s4[2]; d4[3] = s4[3];
    }
    __syncthreads();

    /* ---- step 1: u' = u - w @ S  (per-wave 16x16 tile) ---- */
    v8f uacc;
#pragma unroll
    for (int r = 0; r < 8; ++r)
      uacc[r] = uG[cb + (size_t)(mt * 16 + hi * 8 + r) * DK_ + n0 + nt * 16 + lr];
#pragma unroll
    for (int kk = 0; kk < 8; ++kk) {
      const f16* wr = wG + cb + (size_t)(mt * 16 + lr) * DK_ + kk * 32;
      v16h af = ld_frag_neg(wr + hi * 16, wr + hi * 16 + 8);
      v16h bf;
#pragma unroll
      for (int j = 0; j < 16; ++j)
        bf[j] = (f16)S[(kk * 32 + hi * 16 + j) * SLICE_ + nt * 16 + lr];
      uacc = WMMA(af, bf, uacc);
    }
#pragma unroll
    for (int r = 0; r < 8; ++r)
      up[(mt * 16 + hi * 8 + r) * SLICE_ + nt * 16 + lr] = (f16)uacc[r];

    /* ---- step 2: attn = tril(q kᵀ) (waves 0..3, one 16x16 tile each) ---- */
    if (wv < 4) {
      const int mt2 = wv >> 1, nt2 = wv & 1;
      v8f aacc = {};
#pragma unroll
      for (int kk = 0; kk < 8; ++kk) {
        const f16* qr = qG + cb + (size_t)(mt2 * 16 + lr) * DK_ + kk * 32;
        v16h af = ld_frag(qr + hi * 16, qr + hi * 16 + 8);
        const f16* kr = kt + (nt2 * 16 + lr) * 256 + kk * 32 + hi * 16;
        v16h bf = ld_frag(kr, kr + 8);
        aacc = WMMA(af, bf, aacc);
      }
#pragma unroll
      for (int r = 0; r < 8; ++r) {
        int Mr = mt2 * 16 + hi * 8 + r, Nc = nt2 * 16 + lr;
        at[Mr * 32 + Nc] = (Mr >= Nc) ? (f16)aacc[r] : (f16)0.f;
      }
    }
    __syncthreads();

    /* ---- step 3: o = q @ S + attn @ u' ---- */
    v8f oacc = {};
#pragma unroll
    for (int kk = 0; kk < 8; ++kk) {
      const f16* qr = qG + cb + (size_t)(mt * 16 + lr) * DK_ + kk * 32;
      v16h af = ld_frag(qr + hi * 16, qr + hi * 16 + 8);
      v16h bf;
#pragma unroll
      for (int j = 0; j < 16; ++j)
        bf[j] = (f16)S[(kk * 32 + hi * 16 + j) * SLICE_ + nt * 16 + lr];
      oacc = WMMA(af, bf, oacc);
    }
    {
      const f16* ar = at + (mt * 16 + lr) * 32;
      v16h af = ld_frag(ar + hi * 16, ar + hi * 16 + 8);
      v16h bf;
#pragma unroll
      for (int j = 0; j < 16; ++j)
        bf[j] = up[(hi * 16 + j) * SLICE_ + nt * 16 + lr];
      oacc = WMMA(af, bf, oacc);
    }
#pragma unroll
    for (int r = 0; r < 8; ++r) {
      int lrow = c * CS_ + mt * 16 + hi * 8 + r;
      int d    = n0 + nt * 16 + lr;
      deltaO[((size_t)(b * L_ + lrow)) * 1024 + h * 256 + d] = oacc[r];
    }
    __syncthreads();

    /* ---- step 4: S += kᵀ @ u'  (64 tiles / 8 waves) ---- */
#pragma unroll
    for (int i = 0; i < 8; ++i) {
      const int tt = wv * 8 + i;
      const int mt4 = tt >> 2, nt4 = tt & 3;
      v8f sacc;
#pragma unroll
      for (int r = 0; r < 8; ++r)
        sacc[r] = S[(mt4 * 16 + hi * 8 + r) * SLICE_ + nt4 * 16 + lr];
      v16h af;
#pragma unroll
      for (int j = 0; j < 16; ++j) {
        int kr = hi * 16 + j;
        af[j] = kt[kr * 256 + mt4 * 16 + lr];   /* kᵀ: transposed LDS read */
      }
      v16h bf;
#pragma unroll
      for (int j = 0; j < 16; ++j)
        bf[j] = up[(hi * 16 + j) * SLICE_ + nt4 * 16 + lr];
      sacc = WMMA(af, bf, sacc);
#pragma unroll
      for (int r = 0; r < 8; ++r)
        S[(mt4 * 16 + hi * 8 + r) * SLICE_ + nt4 * 16 + lr] = sacc[r];
    }
    __syncthreads();
  }
}

/* ---------------- FIR short(3)/long(31) depthwise convs ------------------- */

__global__ __launch_bounds__(256) void fir_kernel(const float* __restrict__ v,
    const float* __restrict__ ws, const float* __restrict__ wl,
    float* __restrict__ so, float* __restrict__ loo) {
  size_t i = (size_t)blockIdx.x * 256 + threadIdx.x;
  int ch = (int)(i & 1023);
  size_t bl = i >> 10;
  int l = (int)(bl & (L_ - 1));
  size_t row0 = bl - l;
  float as = 0.f;
#pragma unroll
  for (int j = 0; j < 3; ++j) {
    int ll = l - 2 + j;
    if (ll >= 0) as += v[(row0 + ll) * 1024 + ch] * ws[ch * 3 + j];
  }
  float al = 0.f;
  for (int j = 0; j < 31; ++j) {
    int ll = l - 30 + j;
    if (ll >= 0) al += v[(row0 + ll) * 1024 + ch] * wl[ch * 31 + j];
  }
  so[i] = as; loo[i] = al;
}

/* ---------------- gate input build (concat + head means) -> f16 ----------- */

__global__ __launch_bounds__(256) void gate_in_kernel(const float* __restrict__ hs,
    const float* __restrict__ sh, const float* __restrict__ lo,
    const float* __restrict__ de, f16* __restrict__ gin) {
  const int bl = blockIdx.x;
  const size_t rb = (size_t)bl * 1024;
  for (int idx = threadIdx.x; idx < GIN_; idx += 256) {
    float val;
    if (idx < 1024) {
      val = hs[rb + idx];
    } else if (idx < 1280) {
      int d = idx - 1024;
      val = 0.25f * (sh[rb + d] + sh[rb + 256 + d] + sh[rb + 512 + d] + sh[rb + 768 + d]);
    } else if (idx < 1536) {
      int d = idx - 1280;
      val = 0.25f * (lo[rb + d] + lo[rb + 256 + d] + lo[rb + 512 + d] + lo[rb + 768 + d]);
    } else {
      int d = idx - 1536;
      val = 0.25f * (de[rb + d] + de[rb + 256 + d] + de[rb + 512 + d] + de[rb + 768 + d]);
    }
    gin[(size_t)bl * GIN_ + idx] = (f16)val;
  }
}

/* ---------------- gelu + W2 reduction -> 4 logits per (b,l) --------------- */

__global__ __launch_bounds__(256) void gelu_w2_kernel(const float* __restrict__ hid,
    const float* __restrict__ b1, const float* __restrict__ w2,
    const float* __restrict__ b2, float* __restrict__ logits) {
  const int bl = blockIdx.x, t = threadIdx.x;
  __shared__ float red[256];
  float acc[4] = {0.f, 0.f, 0.f, 0.f};
  for (int idx = t; idx < GH_; idx += 256) {
    float x = hid[(size_t)bl * GH_ + idx] + b1[idx];
    float g = 0.5f * x * (1.f + erff(x * 0.70710678118654752f));
#pragma unroll
    for (int n = 0; n < 4; ++n) acc[n] += g * w2[n * GH_ + idx];
  }
  for (int n = 0; n < 4; ++n) {
    red[t] = acc[n];
    __syncthreads();
    for (int off = 128; off >= 1; off >>= 1) { if (t < off) red[t] += red[t + off]; __syncthreads(); }
    if (t == 0) logits[(size_t)bl * 4 + n] = red[0] + b2[n];
    __syncthreads();
  }
}

/* ---------------- softmax gate + combine + RMSNorm -> o (f16) ------------- */

__global__ __launch_bounds__(256) void combine_kernel(const float* __restrict__ logits,
    const float* __restrict__ logit_temp,
    const float* __restrict__ sh, const float* __restrict__ lo,
    const float* __restrict__ de, const float* __restrict__ vv,
    const float* __restrict__ onw, f16* __restrict__ o16) {
  const int bl = blockIdx.x, t = threadIdx.x;
  const int h = t >> 6, j = t & 63;
  __shared__ float red[256];
  float temp = logf(1.f + expf(logit_temp[0])) + 1e-4f;
  float g0 = logits[(size_t)bl*4+0] / temp, g1 = logits[(size_t)bl*4+1] / temp;
  float g2 = logits[(size_t)bl*4+2] / temp, g3 = logits[(size_t)bl*4+3] / temp;
  float m = fmaxf(fmaxf(g0, g1), fmaxf(g2, g3));
  float e0 = expf(g0-m), e1 = expf(g1-m), e2 = expf(g2-m), e3 = expf(g3-m);
  float s = e0+e1+e2+e3;
  float w0 = e0/s, w1 = e1/s, w2 = e2/s, w3 = e3/s;
  float fl = 0.05f * w3;
  w0 = fmaxf(w0, fl); w1 = fmaxf(w1, fl);
  float s2 = w0+w1+w2+w3;
  w0 /= s2; w1 /= s2; w2 /= s2; w3 /= s2;
  const size_t ib = (size_t)bl * 1024 + h * 256 + j * 4;
  float o[4];
#pragma unroll
  for (int dd = 0; dd < 4; ++dd)
    o[dd] = w0*sh[ib+dd] + w1*lo[ib+dd] + w2*de[ib+dd] + w3*vv[ib+dd];
  red[t] = o[0]*o[0]+o[1]*o[1]+o[2]*o[2]+o[3]*o[3];
  __syncthreads();
  for (int off = 32; off >= 1; off >>= 1) { if (j < off) red[t] += red[t+off]; __syncthreads(); }
  float rms = rsqrtf(red[h*64] / 256.f + 1e-5f);
#pragma unroll
  for (int dd = 0; dd < 4; ++dd)
    o16[ib+dd] = (f16)(o[dd] * rms * onw[j*4+dd]);
}

/* ======================== launch ========================================== */

extern "C" void kernel_launch(void* const* d_in, const int* in_sizes, int n_in,
                              void* d_out, int out_size, void* d_ws, size_t ws_size,
                              hipStream_t stream) {
  (void)in_sizes; (void)n_in; (void)out_size; (void)ws_size;
  const float* hs   = (const float*)d_in[0];
  const float* Wq   = (const float*)d_in[1];
  const float* Wk   = (const float*)d_in[2];
  const float* Wv   = (const float*)d_in[3];
  const float* Wb   = (const float*)d_in[4];
  const float* cq   = (const float*)d_in[5];
  const float* ck   = (const float*)d_in[6];
  const float* cv   = (const float*)d_in[7];
  const float* fws  = (const float*)d_in[8];
  const float* fwl  = (const float*)d_in[9];
  const float* W1   = (const float*)d_in[10];
  const float* b1   = (const float*)d_in[11];
  const float* W2   = (const float*)d_in[12];
  const float* b2   = (const float*)d_in[13];
  const float* ltmp = (const float*)d_in[14];
  const float* onw  = (const float*)d_in[15];
  const float* Wo   = (const float*)d_in[16];

  char* ws = (char*)d_ws;
  /* workspace layout (all sizes multiples of 256B); total ~408 MB */
  size_t off = 0;
  auto take = [&](size_t bytes) { size_t o = off; off += bytes; return o; };
  const size_t o_hs16  = take((size_t)BL_ * D_ * 2);
  const size_t o_wq16  = take((size_t)1024 * 1024 * 2);
  const size_t o_wk16  = take((size_t)1024 * 1024 * 2);
  const size_t o_wv16  = take((size_t)1024 * 1024 * 2);
  const size_t o_w116  = take((size_t)GH_ * GIN_ * 2);
  const size_t o_wo16  = take((size_t)1024 * 1024 * 2);
  const size_t o_qlin  = take((size_t)BL_ * 1024 * 4);  /* later reused: qh (f16) */
  const size_t o_klin  = take((size_t)BL_ * 1024 * 4);  /* later reused: kh (f16) */
  const size_t o_vlin  = take((size_t)BL_ * 1024 * 4);  /* later reused: kb (f16) */
  const size_t o_qc    = take((size_t)BL_ * 1024 * 4);  /* later reused: u  (f32) */
  const size_t o_kc    = take((size_t)BL_ * 1024 * 4);  /* later reused: w  (f16) */
  const size_t o_vc    = take((size_t)BL_ * 1024 * 4);
  const size_t o_beta  = take((size_t)BL_ * 4 * 4);
  const size_t o_vb    = take((size_t)BL_ * 1024 * 4);
  const size_t o_delta = take((size_t)BL_ * 1024 * 4);
  const size_t o_short = take((size_t)BL_ * 1024 * 4);
  const size_t o_long  = take((size_t)BL_ * 1024 * 4);
  const size_t o_gin   = take((size_t)BL_ * GIN_ * 2);
  const size_t o_ghid  = take((size_t)BL_ * GH_ * 4);   /* later reused: o16 (f16) */
  const size_t o_logit = take((size_t)BL_ * 4 * 4);

  f16*   hs16 = (f16*)(ws + o_hs16);
  f16*   wq16 = (f16*)(ws + o_wq16);
  f16*   wk16 = (f16*)(ws + o_wk16);
  f16*   wv16 = (f16*)(ws + o_wv16);
  f16*   w116 = (f16*)(ws + o_w116);
  f16*   wo16 = (f16*)(ws + o_wo16);
  float* qlin = (float*)(ws + o_qlin);
  float* klin = (float*)(ws + o_klin);
  float* vlin = (float*)(ws + o_vlin);
  float* qcb  = (float*)(ws + o_qc);
  float* kcb  = (float*)(ws + o_kc);
  float* vcb  = (float*)(ws + o_vc);
  float* beta = (float*)(ws + o_beta);
  f16*   qh   = (f16*)(ws + o_qlin);
  f16*   kh   = (f16*)(ws + o_klin);
  f16*   kb   = (f16*)(ws + o_vlin);
  float* vb   = (float*)(ws + o_vb);
  float* uB   = (float*)(ws + o_qc);
  f16*   wB   = (f16*)(ws + o_kc);
  float* dlt  = (float*)(ws + o_delta);
  float* shB  = (float*)(ws + o_short);
  float* loB  = (float*)(ws + o_long);
  f16*   gin  = (f16*)(ws + o_gin);
  float* ghid = (float*)(ws + o_ghid);
  float* lgt  = (float*)(ws + o_logit);
  f16*   o16  = (f16*)(ws + o_ghid);

  /* 1) f32 -> f16 conversions */
  auto cvt = [&](const float* src, f16* dst, size_t n) {
    cvt_f16_kernel<<<dim3((unsigned)((n + 255) / 256)), dim3(256), 0, stream>>>(src, dst, n);
  };
  cvt(hs, hs16, (size_t)BL_ * D_);
  cvt(Wq, wq16, (size_t)1024 * 1024);
  cvt(Wk, wk16, (size_t)1024 * 1024);
  cvt(Wv, wv16, (size_t)1024 * 1024);
  cvt(W1, w116, (size_t)GH_ * GIN_);
  cvt(Wo, wo16, (size_t)1024 * 1024);

  /* 2) projections (WMMA GEMMs) + beta */
  dim3 gg(1024 / 128, BL_ / 128);
  gemm_f16_kernel<<<gg, 256, 0, stream>>>(hs16, wq16, qlin, BL_, 1024, 1024);
  gemm_f16_kernel<<<gg, 256, 0, stream>>>(hs16, wk16, klin, BL_, 1024, 1024);
  gemm_f16_kernel<<<gg, 256, 0, stream>>>(hs16, wv16, vlin, BL_, 1024, 1024);
  bproj_kernel<<<dim3(BL_ * 4 / 256), 256, 0, stream>>>(hs, Wb, beta);

  /* 3) causal conv k=4 + SiLU */
  const unsigned nElemBlks = (unsigned)((size_t)BL_ * 1024 / 256);
  conv4_silu_kernel<<<nElemBlks, 256, 0, stream>>>(qlin, cq, qcb);
  conv4_silu_kernel<<<nElemBlks, 256, 0, stream>>>(klin, ck, kcb);
  conv4_silu_kernel<<<nElemBlks, 256, 0, stream>>>(vlin, cv, vcb);

  /* 4) delta-rule preprocessing */
  delta_pre_kernel<<<BL_, 256, 0, stream>>>(qcb, kcb, vcb, beta, qh, kh, kb, vb);

  /* 5) per-chunk Gram + triangular solves (u, w) */
  const unsigned cpLds = 32 * 32 * 4 + 2 * 32 * 256 * 4;   /* 69,632 B */
  chunk_prep_kernel<<<B_ * H_ * NC_, 128, cpLds, stream>>>(kh, kb, vb, uB, wB);

  /* 6) sequential delta scan, dv-sliced (32 WGs; S slice f32 in LDS) */
  const unsigned dsLds = 256 * SLICE_ * 4 + 32 * 256 * 2 + 32 * SLICE_ * 2 + 32 * 32 * 2;
  delta_scan_kernel<<<B_ * H_ * NSLICE_, 256, dsLds, stream>>>(qh, kh, uB, wB, dlt);

  /* 7) FIR convs on v */
  fir_kernel<<<nElemBlks, 256, 0, stream>>>(vcb, fws, fwl, shB, loB);

  /* 8) gate input, gate MLP (WMMA GEMM), logits */
  gate_in_kernel<<<BL_, 256, 0, stream>>>(hs, shB, loB, dlt, gin);
  gemm_f16_kernel<<<gg, 256, 0, stream>>>(gin, w116, ghid, BL_, GH_, GIN_);
  gelu_w2_kernel<<<BL_, 256, 0, stream>>>(ghid, b1, W2, b2, lgt);

  /* 9) softmax gate + combine + RMSNorm */
  combine_kernel<<<BL_, 256, 0, stream>>>(lgt, ltmp, shB, loB, dlt, vcb, onw, o16);

  /* 10) output projection -> d_out (f32) */
  gemm_f16_kernel<<<gg, 256, 0, stream>>>(o16, wo16, (float*)d_out, BL_, D_, 1024);
}
